// Resnet_GNN_33835752358336
// MI455X (gfx1250) — compile-verified, matrix operand order checked by reading, not run on hardware
//
#include <hip/hip_runtime.h>

// ---------------------------------------------------------------------------
// MI455X (gfx1250): ResNet18 + SplineConv GNN.
// All convs and spline node transforms -> bf16 WMMA GEMM (f32 accumulate).
// GEMM: zero-bounds-check inner loop (operands pre-padded by repack kernels),
// double-buffered LDS, one barrier per K-step, global loads for tile i+1
// overlapped with the two v_wmma_f32_16x16x32_bf16 of tile i.
// ---------------------------------------------------------------------------

typedef __attribute__((ext_vector_type(16))) __bf16   v16bf;
typedef __attribute__((ext_vector_type(8)))  float    v8f;
typedef __attribute__((ext_vector_type(4)))  unsigned u32x4;

#define BM 64
#define BN 64
#define BK 32
#define LDT 40                         // LDS row stride (elems), 80B rows, 16B aligned
#define BN_SCALE 0.9999950000374997f   // 1/sqrt(1+1e-5)
#define KS 5

__device__ __forceinline__ __bf16 f2bf(float f) {
  union { float f; unsigned u; } in; in.f = f;
  unsigned r = in.u + 0x7FFFu + ((in.u >> 16) & 1u);   // round-to-nearest-even
  union { unsigned short s; __bf16 b; } out; out.s = (unsigned short)(r >> 16);
  return out.b;
}

// ------------------------- bf16 WMMA GEMM ----------------------------------
// C[M,N] = relu?( scale * A@B + bias[N] + resid[M,N] )
//   A  : [M, K] row-major bf16, M%64==0, K%32==0 (zero-padded by producers)
//   BT : [Npad, K] row-major bf16 (B transposed), Npad = ceil(N/64)*64,
//        rows N..Npad-1 zero-filled by producers.
// 256 threads = 8 waves; block tile 64x64; wave tile 16(M) x 32(N).
__global__ __launch_bounds__(256) void gemm_bf16_wmma(
    const __bf16* __restrict__ A, const __bf16* __restrict__ BT,
    float* __restrict__ C, const float* __restrict__ resid,
    const float* __restrict__ bias,
    int M, int N, int K, float scale, int relu, int fix_neginf)
{
  __shared__ __align__(16) __bf16 As[2][BM][LDT];
  __shared__ __align__(16) __bf16 Bs[2][BN][LDT];
  const int tid  = threadIdx.x;
  const int lane = tid & 31;
  const int wave = tid >> 5;
  const int wm   = wave >> 1;          // 0..3 : M sub-tile
  const int wn   = wave & 1;           // 0..1 : N half (32 cols)
  const int bm   = blockIdx.y * BM;
  const int bn   = blockIdx.x * BN;
  const int l    = lane & 15;
  const int hi   = lane >> 4;

  // staging: each thread moves 8 contiguous bf16 (16B) of A and of BT
  const int sr = (tid * 8) >> 5;       // 0..63
  const int sc = (tid * 8) & 31;       // 0,8,16,24
  const __bf16* Aptr = A  + (size_t)(bm + sr) * K + sc;
  const __bf16* Bptr = BT + (size_t)(bn + sr) * K + sc;

  // preload tile 0
  *(u32x4*)&As[0][sr][sc] = *(const u32x4*)Aptr;
  *(u32x4*)&Bs[0][sr][sc] = *(const u32x4*)Bptr;
  __syncthreads();

  v8f acc0 = {}, acc1 = {};
  int buf = 0;
  for (int k0 = 0; k0 < K; k0 += BK) {
    const int nxt = buf ^ 1;
    // issue global loads for tile i+1 (clamped on last iter: harmless reload)
    const int kn = (k0 + BK < K) ? (k0 + BK) : k0;
    u32x4 na = *(const u32x4*)(Aptr + kn);
    u32x4 nb = *(const u32x4*)(Bptr + kn);
    if (kn + BK < K) {                 // global_prefetch_b8 for tile i+2
      __builtin_prefetch(Aptr + kn + BK, 0, 3);
      __builtin_prefetch(Bptr + kn + BK, 0, 3);
    }
    // fragments of tile i: lane's 16 values = two contiguous 8-elem runs
    union Frag { struct { u32x4 a, b; } q; v16bf v; };
    Frag fa, fb0, fb1;
    const int ar  = wm * 16 + l;
    fa.q.a  = *(const u32x4*)&As[buf][ar][hi * 8];
    fa.q.b  = *(const u32x4*)&As[buf][ar][16 + hi * 8];
    const int br0 = wn * 32 + l;
    fb0.q.a = *(const u32x4*)&Bs[buf][br0][hi * 8];
    fb0.q.b = *(const u32x4*)&Bs[buf][br0][16 + hi * 8];
    const int br1 = br0 + 16;
    fb1.q.a = *(const u32x4*)&Bs[buf][br1][hi * 8];
    fb1.q.b = *(const u32x4*)&Bs[buf][br1][16 + hi * 8];
    acc0 = __builtin_amdgcn_wmma_f32_16x16x32_bf16(false, fa.v, false, fb0.v,
                                                   (short)0, acc0, false, false);
    acc1 = __builtin_amdgcn_wmma_f32_16x16x32_bf16(false, fa.v, false, fb1.v,
                                                   (short)0, acc1, false, false);
    // stage tile i+1 into the other buffer, then one barrier
    *(u32x4*)&As[nxt][sr][sc] = na;
    *(u32x4*)&Bs[nxt][sr][sc] = nb;
    __syncthreads();
    buf = nxt;
  }
  // epilogue: C layout VGPR v -> M = v + 8*hi ; N = lane%16
  #pragma unroll
  for (int t = 0; t < 2; ++t) {
    int n = bn + wn * 32 + t * 16 + l;
    if (n < N) {
      float bv = bias ? bias[n] : 0.f;
      #pragma unroll
      for (int v = 0; v < 8; ++v) {
        int m = bm + wm * 16 + hi * 8 + v;   // always < M (M%64==0)
        float r = (t ? acc1[v] : acc0[v]) * scale + bv;
        if (resid) {
          float rv = resid[(size_t)m * N + n];
          if (fix_neginf && rv == -__builtin_inff()) rv = 0.f;
          r += rv;
        }
        if (relu) r = fmaxf(r, 0.f);
        C[(size_t)m * N + n] = r;
      }
    }
  }
}

// ------------------------- repack / data-movement kernels ------------------

// im2col -> bf16 [M, Kpad]; one block per output row m (uniform SALU decode),
// threads stride over k. k in [Kc,Kpad) zero-filled. Generic src strides
// (handles NCHW network input and NHWC intermediates).
__global__ void im2col_bf16(const float* __restrict__ src, __bf16* __restrict__ dst,
    int Cin, int H, int W, int OW, int KW, int stride, int pad,
    long sn, long scs, long sh, long sw, int Kc, int Kpad, int OHOW)
{
  const int m   = blockIdx.x;
  const int n   = m / OHOW;
  const int rem = m - n * OHOW;
  const int oh  = rem / OW;
  const int ow  = rem - oh * OW;
  const int ih0 = oh * stride - pad;
  const int iw0 = ow * stride - pad;
  __bf16* drow = dst + (long)m * Kpad;
  const float* srow = src + (long)n * sn;
  const int kwc = KW * Cin;
  for (int k = threadIdx.x; k < Kpad; k += blockDim.x) {
    float v = 0.f;
    if (k < Kc) {
      int kh = k / kwc; int r2 = k - kh * kwc;
      int kw = r2 / Cin; int c = r2 - kw * Cin;
      int ih = ih0 + kh, iw = iw0 + kw;
      if (ih >= 0 && ih < H && iw >= 0 && iw < W)
        v = srow[(long)c * scs + (long)ih * sh + (long)iw * sw];
    }
    drow[k] = f2bf(v);
  }
}

// conv weight OIHW f32 -> BT[o][k] bf16, [Cout, Kpad], k=(kh,kw,c), zero pad
__global__ void convert_wconv(const float* __restrict__ w, __bf16* __restrict__ b,
                              int Cout, int Cin, int KH, int KW, int Kc, int Kpad,
                              long total)
{
  long gs = (long)gridDim.x * blockDim.x;
  for (long i = (long)blockIdx.x * blockDim.x + threadIdx.x; i < total; i += gs) {
    int o = (int)(i / Kpad);
    int k = (int)(i - (long)o * Kpad);
    float v = 0.f;
    if (k < Kc) {
      int kh = k / (KW * Cin); int r = k - kh * (KW * Cin);
      int kw = r / Cin;        int c = r - kw * Cin;
      v = w[(((long)o * Cin + c) * KH + kh) * KW + kw];
    }
    b[i] = f2bf(v);
  }
}

// spline weight [KK,cin,cout] f32 -> BT[n][i] bf16, n = k*cout+o, [Npad, cin]
__global__ void convert_splw(const float* __restrict__ w, __bf16* __restrict__ b,
                             int KK, int cin, int cout, int Npad, long total)
{
  long gs = (long)gridDim.x * blockDim.x;
  for (long i = (long)blockIdx.x * blockDim.x + threadIdx.x; i < total; i += gs) {
    int n  = (int)(i / cin);
    int ii = (int)(i - (long)n * cin);
    float v = 0.f;
    if (n < KK * cout) {
      int k = n / cout, o = n - k * cout;
      v = w[((long)k * cin + ii) * cout + o];
    }
    b[i] = f2bf(v);
  }
}

// root [cin,cout] f32 -> BT[o][i] bf16, [Npad, cin], zero pad rows >= cout
__global__ void transpose_cvt(const float* __restrict__ s, __bf16* __restrict__ d,
                              int cin, int cout, int Npad, long total)
{
  long gs = (long)gridDim.x * blockDim.x;
  for (long i = (long)blockIdx.x * blockDim.x + threadIdx.x; i < total; i += gs) {
    int o  = (int)(i / cin);
    int ii = (int)(i - (long)o * cin);
    d[i] = f2bf(o < cout ? s[(long)ii * cout + o] : 0.f);
  }
}

__global__ void cvt_bf16(const float* __restrict__ s, __bf16* __restrict__ d, long n)
{
  long gs = (long)gridDim.x * blockDim.x;
  for (long i = (long)blockIdx.x * blockDim.x + threadIdx.x; i < n; i += gs)
    d[i] = f2bf(s[i]);
}

__global__ void fill_neginf(float* __restrict__ p, long n)
{
  long gs = (long)gridDim.x * blockDim.x;
  for (long i = (long)blockIdx.x * blockDim.x + threadIdx.x; i < n; i += gs)
    p[i] = -__builtin_inff();
}

// 3x3 stride-2 pad-1 maxpool, NHWC
__global__ void maxpool_nhwc(const float* __restrict__ in, float* __restrict__ out,
                             int C, int H, int W, int OH, int OW, long total)
{
  long gs = (long)gridDim.x * blockDim.x;
  for (long i = (long)blockIdx.x * blockDim.x + threadIdx.x; i < total; i += gs) {
    int c = (int)(i % C); long r = i / C;
    int ow = (int)(r % OW); r /= OW;
    int oh = (int)(r % OH); int n = (int)(r / OH);
    float m = -__builtin_inff();
    #pragma unroll
    for (int kh = 0; kh < 3; ++kh) {
      int ih = oh * 2 - 1 + kh;
      if (ih < 0 || ih >= H) continue;
      #pragma unroll
      for (int kw = 0; kw < 3; ++kw) {
        int iw = ow * 2 - 1 + kw;
        if (iw < 0 || iw >= W) continue;
        m = fmaxf(m, in[((long)(n * H + ih) * W + iw) * C + c]);
      }
    }
    out[i] = m;
  }
}

// degree-1 open B-spline basis over [0,1]^2
__global__ void edge_basis_k(const float* __restrict__ attr,
                             float* __restrict__ basis, int* __restrict__ widx, int E)
{
  int e = blockIdx.x * blockDim.x + threadIdx.x;
  if (e >= E) return;
  float v0 = attr[e * 2 + 0] * (KS - 1);
  float v1 = attr[e * 2 + 1] * (KS - 1);
  float l0f = floorf(v0), l1f = floorf(v1);
  float f0 = v0 - l0f, f1 = v1 - l1f;
  int l0 = (int)l0f, l1 = (int)l1f;
  #pragma unroll
  for (int s = 0; s < 4; ++s) {
    int b0 = s & 1, b1 = s >> 1;
    int i0 = min(max(l0 + b0, 0), KS - 1);
    int i1 = min(max(l1 + b1, 0), KS - 1);
    basis[e * 4 + s] = (b0 ? f0 : 1.f - f0) * (b1 ? f1 : 1.f - f1);
    widx[e * 4 + s]  = i0 + i1 * KS;
  }
}

__device__ __forceinline__ void atomicMaxFloat(float* addr, float val)
{
  unsigned* ua = (unsigned*)addr;
  unsigned old = *ua;
  for (;;) {
    float f = __uint_as_float(old);
    if (f >= val) break;
    unsigned assumed = old;
    old = atomicCAS(ua, assumed, __float_as_uint(val));
    if (old == assumed) break;
  }
}

// per (edge, out-channel): msg = sum_s basis_s * Y[src, widx_s*cout+o];
// scatter-max into aggr[dst, o]. cout = 1<<cshift.
__global__ void edge_scatter(const float* __restrict__ Y,
                             const float* __restrict__ basis, const int* __restrict__ widx,
                             const int* __restrict__ srcI, const int* __restrict__ dstI,
                             float* __restrict__ aggr, int cshift, int ldy, long total)
{
  long gs = (long)gridDim.x * blockDim.x;
  const int cmask = (1 << cshift) - 1;
  for (long i = (long)blockIdx.x * blockDim.x + threadIdx.x; i < total; i += gs) {
    int e = (int)(i >> cshift);
    int o = (int)(i & cmask);
    const float* yrow = Y + (long)srcI[e] * ldy;
    float v = basis[e*4+0] * yrow[(widx[e*4+0] << cshift) + o]
            + basis[e*4+1] * yrow[(widx[e*4+1] << cshift) + o]
            + basis[e*4+2] * yrow[(widx[e*4+2] << cshift) + o]
            + basis[e*4+3] * yrow[(widx[e*4+3] << cshift) + o];
    atomicMaxFloat(&aggr[((long)dstI[e] << cshift) + o], v);
  }
}

// ------------------------- host orchestration ------------------------------

static inline int idiv(int a, int b) { return (a + b - 1) / b; }
static inline int rnd(int a, int b)  { return idiv(a, b) * b; }
static inline int gsb(long tot) {
  long b = (tot + 255) / 256;
  if (b < 1) b = 1; if (b > 262144) b = 262144;
  return (int)b;
}
static inline int ilog2(int v) { int s = 0; while ((1 << s) < v) ++s; return s; }

static void launch_gemm(hipStream_t st, const __bf16* A, const __bf16* BT, float* C,
                        const float* resid, const float* bias,
                        int M, int N, int K, float scale, int relu, int fix)
{
  dim3 g(idiv(N, BN), M / BM);
  gemm_bf16_wmma<<<g, 256, 0, st>>>(A, BT, C, resid, bias, M, N, K, scale, relu, fix);
}

static void launch_conv(hipStream_t st, const float* in, int in_nchw,
                        int NBc, int Cin, int H, int W,
                        const float* w, int Cout, int KH, int KW, int stride, int pad,
                        float* out, const float* resid, int relu,
                        __bf16* A_im, __bf16* B_w)
{
  int OH = (H + 2 * pad - KH) / stride + 1;
  int OW = (W + 2 * pad - KW) / stride + 1;
  int Kc = KH * KW * Cin;
  int Kpad = rnd(Kc, BK);
  long M = (long)NBc * OH * OW;
  long wtot = (long)Cout * Kpad;   // Cout is a multiple of 64 for all convs
  convert_wconv<<<gsb(wtot), 256, 0, st>>>(w, B_w, Cout, Cin, KH, KW, Kc, Kpad, wtot);
  long sn, scs, sh, sw;
  if (in_nchw) { sn = (long)Cin * H * W; scs = (long)H * W; sh = W;             sw = 1;   }
  else         { sn = (long)H * W * Cin; scs = 1;           sh = (long)W * Cin; sw = Cin; }
  int bt = Kpad >= 256 ? 256 : (Kpad >= 128 ? 128 : 64);
  im2col_bf16<<<dim3((unsigned)M), bt, 0, st>>>(in, A_im, Cin, H, W, OW, KW,
                                                stride, pad, sn, scs, sh, sw,
                                                Kc, Kpad, OH * OW);
  launch_gemm(st, A_im, B_w, out, resid, nullptr, (int)M, Cout, Kpad, BN_SCALE, relu, 0);
}

static void resblock(hipStream_t st, const float* in, int Cin, int H,
                     const float* w1, const float* w2, const float* wd,
                     int Cout, int stride,
                     float* tmp, float* scbuf, float* out,
                     __bf16* A_im, __bf16* B_w)
{
  int OH = (H + 2 - 3) / stride + 1;
  launch_conv(st, in, 0, 1024, Cin, H, H, w1, Cout, 3, 3, stride, 1, tmp, nullptr, 1, A_im, B_w);
  const float* sc = in;
  if (wd) {
    launch_conv(st, in, 0, 1024, Cin, H, H, wd, Cout, 1, 1, stride, 0, scbuf, nullptr, 0, A_im, B_w);
    sc = scbuf;
  }
  launch_conv(st, tmp, 0, 1024, Cout, OH, OH, w2, Cout, 3, 3, 1, 1, out, sc, 1, A_im, B_w);
}

static void spline_layer(hipStream_t st, const float* X, int cin, int cout,
                         const float* w, const float* root, const float* bias,
                         const int* srcI, const int* dstI, int E,
                         const float* basis, const int* widx,
                         float* out, int relu,
                         __bf16* Xbf, __bf16* Bt, __bf16* rootb, float* Y, float* aggr)
{
  const int NN = 1024, KK = KS * KS;
  const int Nw = KK * cout, NwPad = rnd(Nw, BN);
  cvt_bf16<<<gsb((long)NN * cin), 256, 0, st>>>(X, Xbf, (long)NN * cin);
  long bt = (long)NwPad * cin;
  convert_splw<<<gsb(bt), 256, 0, st>>>(w, Bt, KK, cin, cout, NwPad, bt);
  // Y[NN, KK*cout] = X @ W'  ('nko' einsum for all 25 kernels at once)
  launch_gemm(st, Xbf, Bt, Y, nullptr, nullptr, NN, Nw, cin, 1.f, 0, 0);
  int CoPad = rnd(cout, BN);
  long rt = (long)CoPad * cin;
  transpose_cvt<<<gsb(rt), 256, 0, st>>>(root, rootb, cin, cout, CoPad, rt);
  fill_neginf<<<gsb((long)NN * cout), 256, 0, st>>>(aggr, (long)NN * cout);
  long tot = (long)E * cout;
  edge_scatter<<<gsb(tot), 256, 0, st>>>(Y, basis, widx, srcI, dstI, aggr,
                                         ilog2(cout), Nw, tot);
  // out = X@root + bias + (aggr with -inf -> 0), optional relu
  launch_gemm(st, Xbf, rootb, out, aggr, bias, NN, cout, cin, 1.f, relu, 1);
}

extern "C" void kernel_launch(void* const* d_in, const int* in_sizes, int n_in,
                              void* d_out, int out_size, void* d_ws, size_t ws_size,
                              hipStream_t stream)
{
  (void)n_in; (void)out_size; (void)ws_size;
  const float* x     = (const float*)d_in[0];
  const int*   ei    = (const int*)  d_in[1];
  const float* eattr = (const float*)d_in[2];
  const int E = in_sizes[1] / 2;

  const float* wconv1 = (const float*)d_in[3];
  const float* l1[4]  = { (const float*)d_in[4], (const float*)d_in[5],
                          (const float*)d_in[6], (const float*)d_in[7] };
  const float* l2[5]  = { (const float*)d_in[8],  (const float*)d_in[9], (const float*)d_in[10],
                          (const float*)d_in[11], (const float*)d_in[12] };
  const float* l3[5]  = { (const float*)d_in[13], (const float*)d_in[14], (const float*)d_in[15],
                          (const float*)d_in[16], (const float*)d_in[17] };
  const float* l4[5]  = { (const float*)d_in[18], (const float*)d_in[19], (const float*)d_in[20],
                          (const float*)d_in[21], (const float*)d_in[22] };
  const float* s1w = (const float*)d_in[23]; const float* s1r = (const float*)d_in[24];
  const float* s1b = (const float*)d_in[25];
  const float* s3w = (const float*)d_in[26]; const float* s3r = (const float*)d_in[27];
  const float* s3b = (const float*)d_in[28];
  const float* s2w = (const float*)d_in[29]; const float* s2r = (const float*)d_in[30];
  const float* s2b = (const float*)d_in[31];

  char* base = (char*)d_ws; size_t off = 0;
  auto carve = [&](size_t bytes) -> void* {
    void* p = base + off; off += (bytes + 255) & ~(size_t)255; return p;
  };
  __bf16* A_im  = (__bf16*)carve(41943040ull * 2);  // conv1: 262144 x 160 (Kpad)
  __bf16* B_w   = (__bf16*)carve(2359296ull * 2);   // max 512 x 4608
  float*  act0  = (float*) carve(16777216ull * 4);  // 1024x16x16x64
  float*  p1    = (float*) carve(4194304ull * 4);   // 1024x8x8x64 each
  float*  p2    = (float*) carve(4194304ull * 4);
  float*  p3    = (float*) carve(4194304ull * 4);
  float*  p4    = (float*) carve(4194304ull * 4);
  __bf16* Xbf   = (__bf16*)carve(524288ull * 2);    // 1024x512
  __bf16* rootb = (__bf16*)carve(65536ull * 2);     // 128x512 max
  float*  Ybuf  = (float*) carve(3276800ull * 4);   // 1024x3200
  float*  basis = (float*) carve((size_t)E * 4 * 4);
  int*    widx  = (int*)   carve((size_t)E * 4 * 4);
  float*  aggr  = (float*) carve(131072ull * 4);    // 1024x128
  float*  h1    = (float*) carve(131072ull * 4);    // 1024x128
  float*  h3    = (float*) carve(65536ull * 4);     // 1024x64

  // ---- ResNet18 (NHWC activations; every conv = bf16 WMMA implicit GEMM) ----
  launch_conv(stream, x, /*nchw*/1, 1024, 3, 32, 32, wconv1, 64, 7, 7, 2, 3,
              act0, nullptr, 1, A_im, B_w);                    // -> 16x16x64
  {
    long tot = 1024L * 8 * 8 * 64;
    maxpool_nhwc<<<gsb(tot), 256, 0, stream>>>(act0, p1, 64, 16, 16, 8, 8, tot);
  }
  resblock(stream, p1, 64, 8,  l1[0], l1[1], nullptr, 64, 1,  p2, nullptr, p3, A_im, B_w);
  resblock(stream, p3, 64, 8,  l1[2], l1[3], nullptr, 64, 1,  p2, nullptr, p4, A_im, B_w);
  resblock(stream, p4, 64, 8,  l2[0], l2[1], l2[2],  128, 2,  p2, p1, p3, A_im, B_w);
  resblock(stream, p3, 128, 4, l2[3], l2[4], nullptr,128, 1,  p2, nullptr, p4, A_im, B_w);
  resblock(stream, p4, 128, 4, l3[0], l3[1], l3[2],  256, 2,  p2, p1, p3, A_im, B_w);
  resblock(stream, p3, 256, 2, l3[3], l3[4], nullptr,256, 1,  p2, nullptr, p4, A_im, B_w);
  resblock(stream, p4, 256, 2, l4[0], l4[1], l4[2],  512, 2,  p2, p1, p3, A_im, B_w);
  resblock(stream, p3, 512, 1, l4[3], l4[4], nullptr,512, 1,  p2, nullptr, p4, A_im, B_w);
  const float* feats = p4;  // [1024, 512] (1x1 spatial -> avgpool identity)

  // ---- SplineConv GNN stack ----
  const int* srcI = ei;
  const int* dstI = ei + E;
  edge_basis_k<<<idiv(E, 256), 256, 0, stream>>>(eattr, basis, widx, E);
  spline_layer(stream, feats, 512, 128, s1w, s1r, s1b, srcI, dstI, E,
               basis, widx, h1, /*relu*/0, Xbf, B_w, rootb, Ybuf, aggr);
  spline_layer(stream, h1, 128, 64, s3w, s3r, s3b, srcI, dstI, E,
               basis, widx, h3, /*relu*/1, Xbf, B_w, rootb, Ybuf, aggr);
  spline_layer(stream, h3, 64, 32, s2w, s2r, s2b, srcI, dstI, E,
               basis, widx, (float*)d_out, /*relu*/0, Xbf, B_w, rootb, Ybuf, aggr);
}
